// GINEncoder_41351945125992
// MI455X (gfx1250) — matrix operand order repaired and checked by reading, not compiled, not run on hardware
//
#include <hip/hip_runtime.h>

#define N_NODES    50000
#define N_EDGES    800000
#define DIM        128
#define NUM_GRAPHS 512

typedef __bf16 bf16_t;
typedef bf16_t v16bf __attribute__((ext_vector_type(16)));
typedef float  v8f   __attribute__((ext_vector_type(8)));

union Frag { v16bf v; uint4 u[2]; };

// ---------------- utility kernels ----------------

__global__ void k_zero_f(float* __restrict__ p, int n) {
  int i = blockIdx.x * blockDim.x + threadIdx.x;
  if (i < n) p[i] = 0.0f;
}

__global__ void k_count(const int* __restrict__ batch, float* __restrict__ counts) {
  int i = blockIdx.x * blockDim.x + threadIdx.x;
  if (i < N_NODES) atomicAdd(&counts[batch[i]], 1.0f);
}

__global__ void k_inv(const float* __restrict__ counts, float* __restrict__ inv) {
  int g = blockIdx.x * blockDim.x + threadIdx.x;
  if (g < NUM_GRAPHS) inv[g] = 1.0f / fmaxf(counts[g], 1.0f);
}

// one wave (32 lanes) per edge; each lane owns 4 floats of the 128-wide row
__global__ void k_scatter(const float* __restrict__ x, const int* __restrict__ ei,
                          float* __restrict__ agg) {
  unsigned long long t = (unsigned long long)blockIdx.x * blockDim.x + threadIdx.x;
  int lane = (int)(t & 31u);
  unsigned long long e = t >> 5;
  if (e >= (unsigned long long)N_EDGES) return;
  int src = ei[e];
  int dst = ei[N_EDGES + e];
  const float4 v = *(const float4*)(x + (size_t)src * DIM + lane * 4);
  float* d = agg + (size_t)dst * DIM + lane * 4;
  atomicAdd(d + 0, v.x);
  atomicAdd(d + 1, v.y);
  atomicAdd(d + 2, v.z);
  atomicAdd(d + 3, v.w);
}

// ---------------- fused GIN MLP ----------------
// out = relu((x+agg) @ Wa + ba) @ Wb + bb
// mode 0: write out[row][c] (fp32, [N_NODES,DIM])
// mode 1: atomicAdd(out[batch[row]*DIM + c], val * invc[batch[row]])  (mean pool)
__global__ __launch_bounds__(256)
void k_gin_mlp(const float* __restrict__ xin, const float* __restrict__ agg,
               const float* __restrict__ Wa,  const float* __restrict__ ba,
               const float* __restrict__ Wb,  const float* __restrict__ bb,
               float* __restrict__ out,
               const int* __restrict__ batch, const float* __restrict__ invc,
               int mode) {
  __shared__ __align__(16) bf16_t sWa[DIM * DIM];    // transposed: sWa[n*DIM + k]
  __shared__ __align__(16) bf16_t sWb[DIM * DIM];    // transposed
  __shared__ __align__(16) bf16_t sHin[16 * DIM];    // current 16-row activation tile
  __shared__ __align__(16) bf16_t sHmid[16 * DIM];   // inter-layer activations

  const int tid  = threadIdx.x;
  const int wave = tid >> 5;       // 0..7 -> 16-column output tile
  const int lane = tid & 31;
  const int half = lane >> 4;      // K-half selector per ISA layout
  const int nl   = lane & 15;      // matrix row (A) / col (B,C,D) index
  const int nb   = wave * 16;      // column-tile base

  // load + transpose both weight matrices into LDS as bf16
  for (int i = 0; i < 64; ++i) {
    int idx = tid + 256 * i;       // 0..16383, coalesced over k-major source
    int k = idx >> 7, n = idx & 127;
    sWa[n * DIM + k] = (bf16_t)Wa[idx];
    sWb[n * DIM + k] = (bf16_t)Wb[idx];
  }
  __syncthreads();

  // hoist the per-wave B (weight) fragments into registers: they are invariant
  // across all row-tiles of this block (each wave owns one 16-col tile).
  Frag fBa[4], fBb[4];
#pragma unroll
  for (int i = 0; i < 4; ++i) {
    const int kb = 32 * i;
    const bf16_t* ca = &sWa[(nb + nl) * DIM];
    const bf16_t* cb = &sWb[(nb + nl) * DIM];
    fBa[i].u[0] = *(const uint4*)(ca + kb + 8 * half);
    fBa[i].u[1] = *(const uint4*)(ca + kb + 16 + 8 * half);
    fBb[i].u[0] = *(const uint4*)(cb + kb + 8 * half);
    fBb[i].u[1] = *(const uint4*)(cb + kb + 16 + 8 * half);
  }

  const float biasA = ba[nb + nl];
  const float biasB = bb[nb + nl];
  const int rowBase = blockIdx.x * 64;

  for (int rt = 0; rt < 4; ++rt) {
    const int tileBase = rowBase + rt * 16;

    // cooperative load: hin = x + agg, converted to bf16
    for (int i = 0; i < 8; ++i) {
      int idx = tid + 256 * i;     // 0..2047
      int m = idx >> 7, c = idx & 127;
      int row = tileBase + m;
      float v = 0.0f;
      if (row < N_NODES) v = xin[(size_t)row * DIM + c] + agg[(size_t)row * DIM + c];
      sHin[idx] = (bf16_t)v;
      // hide next tile's global latency behind this tile's WMMA work
      int nrow = row + 16;
      if (rt < 3 && nrow < N_NODES) {
        __builtin_prefetch(&xin[(size_t)nrow * DIM + c], 0, 3);
        __builtin_prefetch(&agg[(size_t)nrow * DIM + c], 0, 3);
      }
    }
    __syncthreads();

    // ---- layer 1: acc = hin @ Wa  (4x v_wmma_f32_16x16x32_bf16) ----
    v8f acc = {0.f, 0.f, 0.f, 0.f, 0.f, 0.f, 0.f, 0.f};
    {
      const bf16_t* arow = &sHin[nl * DIM];
#pragma unroll
      for (int i = 0; i < 4; ++i) {
        const int kb = 32 * i;
        Frag a;
        a.u[0] = *(const uint4*)(arow + kb + 8 * half);
        a.u[1] = *(const uint4*)(arow + kb + 16 + 8 * half);
        acc = __builtin_amdgcn_wmma_f32_16x16x32_bf16(
            false, a.v, false, fBa[i].v, (short)0, acc, false, false);
      }
    }

    // bias + relu, stash inter-layer tile in LDS (bf16)
    for (int r = 0; r < 8; ++r) {
      int m = r + 8 * half;
      float v = fmaxf(acc[r] + biasA, 0.0f);
      sHmid[m * DIM + nb + nl] = (bf16_t)v;
    }
    __syncthreads();

    // ---- layer 2: acc2 = hmid @ Wb ----
    v8f acc2 = {0.f, 0.f, 0.f, 0.f, 0.f, 0.f, 0.f, 0.f};
    {
      const bf16_t* arow = &sHmid[nl * DIM];
#pragma unroll
      for (int i = 0; i < 4; ++i) {
        const int kb = 32 * i;
        Frag a;
        a.u[0] = *(const uint4*)(arow + kb + 8 * half);
        a.u[1] = *(const uint4*)(arow + kb + 16 + 8 * half);
        acc2 = __builtin_amdgcn_wmma_f32_16x16x32_bf16(
            false, a.v, false, fBb[i].v, (short)0, acc2, false, false);
      }
    }

    // ---- output ----
    if (mode == 0) {
      for (int r = 0; r < 8; ++r) {
        int row = tileBase + r + 8 * half;
        if (row < N_NODES) out[(size_t)row * DIM + nb + nl] = acc2[r] + biasB;
      }
    } else {
      for (int r = 0; r < 8; ++r) {
        int row = tileBase + r + 8 * half;
        if (row < N_NODES) {
          int g = batch[row];
          atomicAdd(&out[(size_t)g * DIM + nb + nl], (acc2[r] + biasB) * invc[g]);
        }
      }
    }
    __syncthreads();  // protect sHin/sHmid before next row-tile
  }
}

// ---------------- launch ----------------

extern "C" void kernel_launch(void* const* d_in, const int* in_sizes, int n_in,
                              void* d_out, int out_size, void* d_ws, size_t ws_size,
                              hipStream_t stream) {
  const float* x   = (const float*)d_in[0];
  const int*   ei  = (const int*)d_in[1];
  const int*   bat = (const int*)d_in[2];
  const float* W1a = (const float*)d_in[3];
  const float* b1a = (const float*)d_in[4];
  const float* W1b = (const float*)d_in[5];
  const float* b1b = (const float*)d_in[6];
  const float* W2a = (const float*)d_in[7];
  const float* b2a = (const float*)d_in[8];
  const float* W2b = (const float*)d_in[9];
  const float* b2b = (const float*)d_in[10];
  float* out = (float*)d_out;

  // workspace layout (floats)
  float* ws     = (float*)d_ws;
  float* agg    = ws;                         // N_NODES*DIM
  float* h1     = ws + (size_t)N_NODES * DIM; // N_NODES*DIM
  float* counts = h1 + (size_t)N_NODES * DIM; // NUM_GRAPHS
  float* invc   = counts + NUM_GRAPHS;        // NUM_GRAPHS

  const int ND = N_NODES * DIM;               // 6,400,000
  const int mlpBlocks = (N_NODES + 63) / 64;  // 782
  const int scatBlocks = (int)(((unsigned long long)N_EDGES * 32 + 255) / 256);

  // zero agg, output, counts
  k_zero_f<<<(ND + 255) / 256, 256, 0, stream>>>(agg, ND);
  k_zero_f<<<(NUM_GRAPHS * DIM + 255) / 256, 256, 0, stream>>>(out, NUM_GRAPHS * DIM);
  k_zero_f<<<(NUM_GRAPHS + 255) / 256, 256, 0, stream>>>(counts, NUM_GRAPHS);

  // graph sizes -> inverse counts
  k_count<<<(N_NODES + 255) / 256, 256, 0, stream>>>(bat, counts);
  k_inv<<<(NUM_GRAPHS + 255) / 256, 256, 0, stream>>>(counts, invc);

  // layer 1
  k_scatter<<<scatBlocks, 256, 0, stream>>>(x, ei, agg);
  k_gin_mlp<<<mlpBlocks, 256, 0, stream>>>(x, agg, W1a, b1a, W1b, b1b,
                                           h1, nullptr, nullptr, 0);

  // layer 2 (+ fused mean pool)
  k_zero_f<<<(ND + 255) / 256, 256, 0, stream>>>(agg, ND);
  k_scatter<<<scatBlocks, 256, 0, stream>>>(h1, ei, agg);
  k_gin_mlp<<<mlpBlocks, 256, 0, stream>>>(h1, agg, W2a, b2a, W2b, b2b,
                                           out, bat, invc, 1);
}
